// LSTMCell_48507360641621
// MI455X (gfx1250) — matrix operand order repaired and checked by reading, not compiled, not run on hardware
//
#include <hip/hip_runtime.h>
#include <math.h>

typedef __attribute__((ext_vector_type(16))) _Float16 v16h;
typedef __attribute__((ext_vector_type(8)))  _Float16 v8h;
typedef __attribute__((ext_vector_type(8)))  float    v8f;
typedef __attribute__((ext_vector_type(4)))  float    v4f;

#define WMMA_F32_F16(a, b, c) \
  __builtin_amdgcn_wmma_f32_16x16x32_f16(false, (a), false, (b), (short)0, (c), false, false)

// ---------------- workspace layout (bytes) ----------------
// xh   : f16 copy of x            64*512*256*2 = 16,777,216
// Wxh  : f16 copy of Wx           4*256*256*2  =    524,288
// Whh  : f16 copy of Wh           4*256*256*2  =    524,288
// bias : f32 bx+bh                1024*4       =      4,096
// xgf  : f32 WMMA C-fragments, fragment-major:
//        [ (s*4+bt)*64 + ntile ][ lane*8 + j ]  = 512*64*1024*4 = 134,217,728
static constexpr size_t OFF_XH   = 0;
static constexpr size_t OFF_WXH  = 16777216;
static constexpr size_t OFF_WHH  = OFF_WXH + 524288;
static constexpr size_t OFF_BIAS = OFF_WHH + 524288;
static constexpr size_t OFF_XG   = OFF_BIAS + 4096;

__device__ __forceinline__ float sigmoidf_(float x) {
  return 1.0f / (1.0f + __expf(-x));
}

// build v16h A-fragment from two 16-byte chunks (contiguous K layout)
__device__ __forceinline__ v16h mk_afrag(const _Float16* p0, const _Float16* p1) {
  v8h lo = *(const v8h*)p0;
  v8h hi = *(const v8h*)p1;
  return __builtin_shufflevector(lo, hi, 0, 1, 2, 3, 4, 5, 6, 7,
                                 8, 9, 10, 11, 12, 13, 14, 15);
}

// ---------------- prep kernels ----------------
__global__ void f32_to_f16_k(const float* __restrict__ src,
                             _Float16* __restrict__ dst, int n) {
  int i = blockIdx.x * blockDim.x + threadIdx.x;
  int stride = gridDim.x * blockDim.x;
  for (; i < n; i += stride) dst[i] = (_Float16)src[i];
}

__global__ void bias_sum_k(const float* __restrict__ bx,
                           const float* __restrict__ bh,
                           float* __restrict__ bias) {
  int i = threadIdx.x;  // 1024 threads, 1024 elems
  bias[i] = bx[i] + bh[i];
}

// ---------------- kernel 1: xg = x @ Wx^T + (bx+bh) ----------------
// grid (8, 2048), block 256 (8 waves). wave -> one 16x16 tile.
// GEMM row r = s*64 + b; col n = gate*256 + o.
// Output stored as raw per-lane WMMA C-fragments (v8f, 32B contiguous per
// lane, 1KB contiguous per wave) so the scan can reload accumulators with a
// single vector load: frag_idx = blockIdx.y*64 + ntile.
__global__ __launch_bounds__(256) void xg_gemm_k(
    const _Float16* __restrict__ xh,   // [B][S][I] f16
    const _Float16* __restrict__ Wxh,  // [4*O][I] f16
    const float* __restrict__ bias,    // [1024]
    float* __restrict__ xgf) {         // fragment-major C blobs
  const int wave = threadIdx.x >> 5;
  const int lane = threadIdx.x & 31;
  const int lr = lane & 15;            // A-row / B-col / D-col within tile
  const int kh = (lane >> 4) * 16;     // A contiguous-K offset
  const int ks = (lane >> 4) * 16;     // B contiguous-K offset

  const int ntile = blockIdx.x * 8 + wave;      // 0..63
  const int nb = ntile * 16;                    // col base
  const int rowbase = blockIdx.y * 16;          // row base (0..32752)
  const int r = rowbase + lr;
  const int s = r >> 6, b = r & 63;

  const _Float16* ap = xh + ((size_t)(b * 512 + s)) * 256 + kh;
  const _Float16* bp = Wxh + (size_t)(nb + lr) * 256 + ks;

  const float bv = bias[nb + lr];
  v8f acc;
#pragma unroll
  for (int j = 0; j < 8; ++j) acc[j] = bv;

#pragma unroll
  for (int k = 0; k < 8; ++k) {
    const int kb = k * 32;
    v16h a = mk_afrag(ap + kb, ap + kb + 8);
    v16h bm = *(const v16h*)(bp + kb);
    acc = WMMA_F32_F16(a, bm, acc);
  }

  // fragment-major store: one contiguous 32B per lane
  const size_t frag = (size_t)blockIdx.y * 64 + ntile;
  *(v8f*)(xgf + frag * 256 + lane * 8) = acc;
}

// ---------------- kernel 2: recurrent scan ----------------
// grid 4 (batch tiles of 16 rows), block 1024 (32 waves).
// Per step: g = xg[s] + h @ Wh^T (WMMA, xg fragment loaded as C), then gates.
// xg fragments for step s+1 are prefetched into registers during step s so
// their load latency hides behind the WMMA chain + elementwise phase.
#define HS 272        // h_sh row stride in halves (256 + 16 pad, 16B aligned rows)
#define GS 1032       // g_sh row stride in halves

__global__ __launch_bounds__(1024) void lstm_scan_k(
    const float* __restrict__ xgf,     // fragment-major C blobs
    const _Float16* __restrict__ Whh,  // [1024][256] f16
    const float* __restrict__ h0,      // [B][256]
    const float* __restrict__ c0,      // [B][256]
    float* __restrict__ out) {         // h_seq | hT | cT
  __shared__ _Float16 h_sh[16 * HS];   // h_t as f16 (A operand source)
  __shared__ _Float16 g_sh[16 * GS];   // gate pre-activations (f16 staging)

  const int t = threadIdx.x;
  const int bt = blockIdx.x;           // batch tile 0..3
  const int wave = t >> 5;
  const int lane = t & 31;
  const int lr = lane & 15;
  const int kh = (lane >> 4) * 16;
  const int ks = (lane >> 4) * 16;
  const int rj = (lane >> 4) * 8;

  // elementwise mapping: 1024 threads x 4 elems = 16 rows x 256 cols
  const int erow = t >> 6;             // 0..15
  const int ecol = (t & 63) * 4;       // 0,4,...,252

  // ---- init h, c ----
  float c[4], hv[4];
  {
    const float* h0p = h0 + (size_t)(bt * 16 + erow) * 256 + ecol;
    const float* c0p = c0 + (size_t)(bt * 16 + erow) * 256 + ecol;
#pragma unroll
    for (int j = 0; j < 4; ++j) {
      c[j] = c0p[j];
      float h = h0p[j];
      hv[j] = h;
      h_sh[erow * HS + ecol + j] = (_Float16)h;
    }
  }
  __syncthreads();

  const int nb0 = wave * 32;           // this wave's 2 tiles: cols nb0, nb0+16
  const _Float16* bp0 = Whh + (size_t)(nb0 + lr) * 256 + ks;
  const _Float16* bp1 = Whh + (size_t)(nb0 + 16 + lr) * 256 + ks;
  const _Float16* asrc = &h_sh[lr * HS + kh];
  // per-step C-fragment source: tiles (2*wave) and (2*wave+1) of row-tile
  // (s*4 + bt); advance by 256 tiles (256KB) per step
  const size_t STEP = (size_t)4 * 64 * 256;
  const float* xfp0 = xgf + ((size_t)bt * 64 + wave * 2) * 256 + lane * 8;

  // prime the pipeline: step 0 fragments
  v8f nxt0 = *(const v8f*)(xfp0);
  v8f nxt1 = *(const v8f*)(xfp0 + 256);

  for (int s = 0; s < 512; ++s) {
    v8f acc0 = nxt0;
    v8f acc1 = nxt1;

    // prefetch step s+1 fragments (guarded re-read of step 0 on last iter);
    // issued before the WMMA chain so latency hides behind compute+barriers
    const float* pf = (s < 511) ? (xfp0 + (size_t)(s + 1) * STEP) : xfp0;
    nxt0 = *(const v8f*)(pf);
    nxt1 = *(const v8f*)(pf + 256);

#pragma unroll
    for (int k = 0; k < 8; ++k) {
      const int kb = k * 32;
      v16h a = mk_afrag(asrc + kb, asrc + kb + 8);   // LDS (ds_load_b128 x2)
      v16h b0 = *(const v16h*)(bp0 + kb);            // Wh from L2
      v16h b1 = *(const v16h*)(bp1 + kb);
      acc0 = WMMA_F32_F16(a, b0, acc0);
      acc1 = WMMA_F32_F16(a, b1, acc1);
    }

    // stage gate pre-activations for cross-wave elementwise phase
#pragma unroll
    for (int j = 0; j < 8; ++j) {
      g_sh[(j + rj) * GS + nb0 + lr]      = (_Float16)acc0[j];
      g_sh[(j + rj) * GS + nb0 + 16 + lr] = (_Float16)acc1[j];
    }
    __syncthreads();

    // elementwise LSTM cell update; c stays in registers
    v4f h4;
#pragma unroll
    for (int j = 0; j < 4; ++j) {
      const int gbase = erow * GS + ecol + j;
      float f  = sigmoidf_((float)g_sh[gbase]);
      float ig = sigmoidf_((float)g_sh[gbase + 256]);
      float gt = tanhf((float)g_sh[gbase + 512]);
      float og = sigmoidf_((float)g_sh[gbase + 768]);
      float cn = c[j] * f + ig * gt;
      c[j] = cn;
      float h = og * tanhf(cn);
      hv[j] = h;
      h4[j] = h;
      h_sh[erow * HS + ecol + j] = (_Float16)h;  // feed next step's A
    }
    // h_seq: write-once stream; nontemporal so it doesn't evict Wh/xg in L2
    __builtin_nontemporal_store(
        h4, (v4f*)(out + ((size_t)(bt * 16 + erow) * 512 + s) * 256 + ecol));
    __syncthreads();
  }

  // final hT, cT
  float* hT = out + (size_t)64 * 512 * 256;
  float* cT = hT + 64 * 256;
  const size_t o = (size_t)(bt * 16 + erow) * 256 + ecol;
#pragma unroll
  for (int j = 0; j < 4; ++j) {
    hT[o + j] = hv[j];
    cT[o + j] = c[j];
  }
}

// ---------------- host ----------------
extern "C" void kernel_launch(void* const* d_in, const int* in_sizes, int n_in,
                              void* d_out, int out_size, void* d_ws, size_t ws_size,
                              hipStream_t stream) {
  // setup_inputs order: x, h0, c0, Wh, bh, Wx, bx
  const float* x  = (const float*)d_in[0];
  const float* h0 = (const float*)d_in[1];
  const float* c0 = (const float*)d_in[2];
  const float* Wh = (const float*)d_in[3];
  const float* bh = (const float*)d_in[4];
  const float* Wx = (const float*)d_in[5];
  const float* bx = (const float*)d_in[6];

  char* ws = (char*)d_ws;
  _Float16* xh   = (_Float16*)(ws + OFF_XH);
  _Float16* Wxh  = (_Float16*)(ws + OFF_WXH);
  _Float16* Whh  = (_Float16*)(ws + OFF_WHH);
  float*    bias = (float*)   (ws + OFF_BIAS);
  float*    xgf  = (float*)   (ws + OFF_XG);

  f32_to_f16_k<<<4096, 256, 0, stream>>>(x,  xh,  64 * 512 * 256);
  f32_to_f16_k<<<512,  256, 0, stream>>>(Wx, Wxh, 4 * 256 * 256);
  f32_to_f16_k<<<512,  256, 0, stream>>>(Wh, Whh, 4 * 256 * 256);
  bias_sum_k<<<1, 1024, 0, stream>>>(bx, bh, bias);

  // xg GEMM: 2048 M-tiles x 64 N-tiles; 8 waves/block -> one tile each
  xg_gemm_k<<<dim3(8, 2048), 256, 0, stream>>>(xh, Wxh, bias, xgf);

  // recurrent scan: 4 persistent workgroups (one per 16-row batch tile)
  lstm_scan_k<<<4, 1024, 0, stream>>>(xgf, Whh, h0, c0, (float*)d_out);
}
